// SutraV061_31318901522951
// MI455X (gfx1250) — matrix operand order repaired and checked by reading, not compiled, not run on hardware
//
#include <hip/hip_runtime.h>
#include <math.h>

typedef __bf16 v16bf __attribute__((ext_vector_type(16)));
typedef __bf16 v2bf  __attribute__((ext_vector_type(2)));
typedef float  v8f   __attribute__((ext_vector_type(8)));

#define OP_NONE        0
#define OP_GELU        1
#define OP_SILU        2
#define OP_SIGMOID     3
#define OP_SOFTPLUS    4
#define OP_SOFTPLUS01  5

#define BTOK 512      // B*T tokens
#define DD   768
#define FFD  1536
#define NST  6
#define NSL  8
#define VV   50257
#define LDSW 20       // dwords per LDS tile row (32 halves + 8 pad)

__device__ __forceinline__ unsigned pack2(float a, float b) {
  v2bf t; t.x = (__bf16)a; t.y = (__bf16)b;     // hw f32->bf16 converts
  return __builtin_bit_cast(unsigned, t);
}
__device__ __forceinline__ float sigmoidf_(float x) { return 1.f / (1.f + __expf(-x)); }
__device__ __forceinline__ float softplusf_(float x) { return x > 20.f ? x : log1pf(__expf(x)); }
__device__ __forceinline__ float geluf_(float x) {
  float u = 0.7978845608028654f * (x + 0.044715f * x * x * x);
  return 0.5f * x * (1.f + tanhf(u));
}

// ---------------------------------------------------------------------------
// Stage one 64x32 A tile + 64x32(B^T layout) tile into LDS as packed bf16 pairs.
// fast == true: no bounds checks, unconditional coalesced loads, packed b32 stores.
// ---------------------------------------------------------------------------
__device__ __forceinline__ void stage_tile(
    const float* __restrict__ A, const float* __restrict__ Bm,
    int M, int N, int K, int lda, int ldb,
    int m0, int n0, int k0, int transB, bool fast,
    unsigned* As32, unsigned* Bs32, int tid) {
  int kp = tid & 15, r0 = tid >> 4;   // kp: k-pair 0..15, r0: row 0..7
  if (fast) {
    {
      const float* Ap = A + (long)(m0 + r0) * lda + (k0 + kp * 2);
#pragma unroll
      for (int i = 0; i < 8; ++i) {
        long o = (long)(i * 8) * lda;
        As32[(r0 + i * 8) * LDSW + kp] = pack2(Ap[o], Ap[o + 1]);
      }
    }
    if (transB) {
      const float* Bp = Bm + (long)(n0 + r0) * ldb + (k0 + kp * 2);
#pragma unroll
      for (int i = 0; i < 8; ++i) {
        long o = (long)(i * 8) * ldb;
        Bs32[(r0 + i * 8) * LDSW + kp] = pack2(Bp[o], Bp[o + 1]);
      }
    } else {
      int n = tid & 63, kp0 = (tid >> 6) * 8;
      const float* Bp = Bm + (long)(k0 + kp0 * 2) * ldb + (n0 + n);
#pragma unroll
      for (int i = 0; i < 8; ++i) {
        long o = (long)(2 * i) * ldb;
        Bs32[n * LDSW + kp0 + i] = pack2(Bp[o], Bp[o + ldb]);
      }
    }
  } else {
    {
#pragma unroll
      for (int i = 0; i < 8; ++i) {
        int m = r0 + i * 8, gm = m0 + m, gk = k0 + kp * 2;
        float x0 = (gm < M && gk < K)     ? A[(long)gm * lda + gk]     : 0.f;
        float x1 = (gm < M && gk + 1 < K) ? A[(long)gm * lda + gk + 1] : 0.f;
        As32[m * LDSW + kp] = pack2(x0, x1);
      }
    }
    if (transB) {
#pragma unroll
      for (int i = 0; i < 8; ++i) {
        int n = r0 + i * 8, gn = n0 + n, gk = k0 + kp * 2;
        float x0 = (gn < N && gk < K)     ? Bm[(long)gn * ldb + gk]     : 0.f;
        float x1 = (gn < N && gk + 1 < K) ? Bm[(long)gn * ldb + gk + 1] : 0.f;
        Bs32[n * LDSW + kp] = pack2(x0, x1);
      }
    } else {
      int n = tid & 63, kp0 = (tid >> 6) * 8, gn = n0 + n;
#pragma unroll
      for (int i = 0; i < 8; ++i) {
        int gk = k0 + (kp0 + i) * 2;
        float x0 = (gn < N && gk < K)     ? Bm[(long)gk * ldb + gn]       : 0.f;
        float x1 = (gn < N && gk + 1 < K) ? Bm[(long)(gk + 1) * ldb + gn] : 0.f;
        Bs32[n * LDSW + kp0 + i] = pack2(x0, x1);
      }
    }
  }
}

// 2x2 of v_wmma_f32_16x16x32_bf16 on one staged tile (per wave 32x32 patch)
__device__ __forceinline__ void tile_mma(const unsigned* As32, const unsigned* Bs32,
                                         int wm, int wn, int lh, int hi,
                                         v8f (&acc)[2][2]) {
  union Frag { v16bf v; unsigned u[8]; };
  Frag fa[2], fb[2];
  // A fragment (16x32): lanes 0-15 rows M; halves 0..15 = K hi*16..hi*16+15
#pragma unroll
  for (int ti = 0; ti < 2; ++ti) {
    int base = (wm + ti * 16 + lh) * LDSW + hi * 8;
#pragma unroll
    for (int j = 0; j < 4; ++j) {
      fa[ti].u[j]     = As32[base + j];
      fa[ti].u[4 + j] = As32[base + 4 + j];
    }
  }
  // B fragment (32x16): lane -> column; lanes<16 K=0..15, lanes>=16 K=16..31
#pragma unroll
  for (int tj = 0; tj < 2; ++tj) {
    int base = (wn + tj * 16 + lh) * LDSW + hi * 8;
#pragma unroll
    for (int j = 0; j < 8; ++j) fb[tj].u[j] = Bs32[base + j];
  }
#pragma unroll
  for (int ti = 0; ti < 2; ++ti)
#pragma unroll
    for (int tj = 0; tj < 2; ++tj)
      acc[ti][tj] = __builtin_amdgcn_wmma_f32_16x16x32_bf16(
          false, fa[ti].v, false, fb[tj].v, (short)0, acc[ti][tj], false, false);
}

// ---------------------------------------------------------------------------
// Generic bf16-WMMA GEMM: C = op(alpha * A*B + bias)
// A: MxK f32 row-major; B: KxN f32 row-major (or NxK if transB).
// Batched via blockIdx.z. 128 threads (4 waves), 64x64 tile, BK=32,
// double-buffered LDS (one barrier per K-step, loads overlap WMMA).
// ---------------------------------------------------------------------------
__global__ void __launch_bounds__(128) wmma_gemm(
    const float* __restrict__ A, const float* __restrict__ Bm,
    const float* __restrict__ bias, float* __restrict__ C,
    int M, int N, int K, int lda, int ldb, int ldc,
    long sA, long sB, long sC, long sBias, int transB, int op, float alpha) {
  int z = blockIdx.z;
  A += (long)z * sA; Bm += (long)z * sB; C += (long)z * sC;
  const float* bp = bias ? bias + (long)z * sBias : nullptr;
  int m0 = blockIdx.y * 64, n0 = blockIdx.x * 64;

  __shared__ unsigned AsB[2][64 * LDSW];
  __shared__ unsigned BsB[2][64 * LDSW];

  int tid = threadIdx.x, lane = tid & 31, w = tid >> 5;
  int wm = (w >> 1) * 32, wn = (w & 1) * 32;
  int lh = lane & 15, hi = lane >> 4;

  bool fullMN = (m0 + 64 <= M) && (n0 + 64 <= N);
  v8f acc[2][2] = {};

  int nk = (K + 31) >> 5;
  stage_tile(A, Bm, M, N, K, lda, ldb, m0, n0, 0, transB,
             fullMN && (32 <= K), AsB[0], BsB[0], tid);
  int cur = 0;
  for (int it = 0; it < nk; ++it) {
    __syncthreads();
    if (it + 1 < nk) {
      int k0 = (it + 1) * 32;
      stage_tile(A, Bm, M, N, K, lda, ldb, m0, n0, k0, transB,
                 fullMN && (k0 + 32 <= K), AsB[cur ^ 1], BsB[cur ^ 1], tid);
    }
    tile_mma(AsB[cur], BsB[cur], wm, wn, lh, hi, acc);
    cur ^= 1;
  }

#pragma unroll
  for (int ti = 0; ti < 2; ++ti)
    for (int tj = 0; tj < 2; ++tj)
#pragma unroll
      for (int r = 0; r < 8; ++r) {
        int gm = m0 + wm + ti * 16 + r + hi * 8;   // C/D: VGPR r, lanes>=16 -> M+8
        int gn = n0 + wn + tj * 16 + lh;
        if (fullMN || (gm < M && gn < N)) {
          float vv = acc[ti][tj][r] * alpha;
          if (bp) vv += bp[gn];
          switch (op) {
            case OP_GELU:       vv = geluf_(vv); break;
            case OP_SILU:       vv = vv * sigmoidf_(vv); break;
            case OP_SIGMOID:    vv = sigmoidf_(vv); break;
            case OP_SOFTPLUS:   vv = softplusf_(vv); break;
            case OP_SOFTPLUS01: vv = softplusf_(vv) + 0.1f; break;
          }
          C[(long)gm * ldc + gn] = vv;
        }
      }
}

// ---------------------------------------------------------------------------
__device__ __forceinline__ float blockReduce256(float v, float* s) {
  int t = threadIdx.x;
  s[t] = v; __syncthreads();
  for (int o = 128; o > 0; o >>= 1) { if (t < o) s[t] += s[t + o]; __syncthreads(); }
  float r = s[0]; __syncthreads();
  return r;
}

__global__ void k_fill(float* p, int n, float v) {
  int i = blockIdx.x * blockDim.x + threadIdx.x;
  if (i < n) p[i] = v;
}
__global__ void k_copy(float* dst, const float* src, int n) {
  int i = blockIdx.x * blockDim.x + threadIdx.x;
  if (i < n) dst[i] = src[i];
}
__global__ void k_pi_init(float* pi) {
  int bt = blockIdx.x * blockDim.x + threadIdx.x;
  if (bt < BTOK) pi[bt * NST + 2] = 1.f;
}
__global__ void k_embed(const int* x, const float* emb, const float* pos, float* h) {
  int bt = blockIdx.x, t = bt & 255;
  for (int d = threadIdx.x; d < DD; d += 256)
    h[(long)bt * DD + d] = emb[(long)x[bt] * DD + d] + pos[(long)t * DD + d];
}
// pe MLP: gb12[p][0:1536] = silu(pe[p]@m1+mb1)@m2 + mb2   (one block per p)
__global__ void k_pe_mlp(const float* pe, const float* m1, const float* mb1,
                         const float* m2, const float* mb2, float* gb12) {
  __shared__ float hh[128];
  int p = blockIdx.x, tid = threadIdx.x;
  if (tid < 128) {
    float a = mb1[tid];
    for (int kk = 0; kk < 64; ++kk) a += pe[p * 64 + kk] * m1[kk * 128 + tid];
    hh[tid] = a * sigmoidf_(a);
  }
  __syncthreads();
  for (int j = tid; j < 2 * DD; j += 128) {
    float a = mb2[j];
    for (int i = 0; i < 128; ++i) a += hh[i] * m2[i * (2 * DD) + j];
    gb12[(long)p * (2 * DD) + j] = a;
  }
}
// PCN finish: out = g * (sigmoid(gpre) * LN(x)) + beta
__global__ void k_pcn_finish(const float* x, const float* gpre, const float* g12, int p,
                             const float* lw, const float* lb, float* out) {
  __shared__ float red[256];
  int bt = blockIdx.x, tid = threadIdx.x;
  const float* xr = x + (long)bt * DD;
  float s = 0;
  for (int d = tid; d < DD; d += 256) s += xr[d];
  float mean = blockReduce256(s, red) * (1.f / DD);
  float vs = 0;
  for (int d = tid; d < DD; d += 256) { float t = xr[d] - mean; vs += t * t; }
  float rstd = rsqrtf(blockReduce256(vs, red) * (1.f / DD) + 1e-5f);
  const float* gg = g12 + (long)p * (2 * DD);
  for (int d = tid; d < DD; d += 256) {
    float hh = sigmoidf_(gpre[(long)bt * DD + d]) * ((xr[d] - mean) * rstd * lw[d] + lb[d]);
    out[(long)bt * DD + d] = gg[d] * hh + gg[DD + d];
  }
}
__global__ void k_layernorm(const float* x, const float* wgt, const float* bia, float* out) {
  __shared__ float red[256];
  int bt = blockIdx.x, tid = threadIdx.x;
  const float* xr = x + (long)bt * DD;
  float s = 0;
  for (int d = tid; d < DD; d += 256) s += xr[d];
  float mean = blockReduce256(s, red) * (1.f / DD);
  float vs = 0;
  for (int d = tid; d < DD; d += 256) { float t = xr[d] - mean; vs += t * t; }
  float rstd = rsqrtf(blockReduce256(vs, red) * (1.f / DD) + 1e-5f);
  for (int d = tid; d < DD; d += 256)
    out[(long)bt * DD + d] = (xr[d] - mean) * rstd * wgt[d] + bia[d];
}
// transition softmax + pi evolution: pi_ev[r] = sum_s pi[s]*softmax_r(tr[s][:])
__global__ void k_trans_pi(const float* trout, const float* pi, float* pi_ev) {
  int bt = blockIdx.x * blockDim.x + threadIdx.x;
  if (bt >= BTOK) return;
  const float* tr = trout + (long)bt * 36;
  const float* pr = pi + (long)bt * NST;
  float ev[NST] = {0, 0, 0, 0, 0, 0};
  for (int s = 0; s < NST; ++s) {
    float m = -1e30f;
    for (int r = 0; r < NST; ++r) m = fmaxf(m, tr[s * NST + r]);
    float e[NST], sum = 0;
    for (int r = 0; r < NST; ++r) { e[r] = __expf(tr[s * NST + r] - m); sum += e[r]; }
    float wgt = pr[s] / sum;
    for (int r = 0; r < NST; ++r) ev[r] += wgt * e[r];
  }
  for (int r = 0; r < NST; ++r) pi_ev[(long)bt * NST + r] = ev[r];
}
__global__ void k_stage_mix(const float* outs, const float* pi, float* stage_out) {
  __shared__ float ps[NST];
  int bt = blockIdx.x, tid = threadIdx.x;
  if (tid < NST) ps[tid] = pi[(long)bt * NST + tid];
  __syncthreads();
  for (int d = tid; d < DD; d += 256) {
    float acc = 0;
    for (int s = 0; s < NST; ++s) acc += ps[s] * outs[((long)s * BTOK + bt) * DD + d];
    stage_out[(long)bt * DD + d] = acc;
  }
}
// pi_new = pi_ev * softmax(ev/0.5); normalize; top2 re-normalize; emit g3,g4
__global__ void k_pi_update(const float* pi_ev, const float* evid, float* pi,
                            float* g3, float* g4) {
  int bt = blockIdx.x * blockDim.x + threadIdx.x;
  if (bt >= BTOK) return;
  const float* e = evid + (long)bt * NST;
  float m = -1e30f;
  for (int r = 0; r < NST; ++r) m = fmaxf(m, e[r] * 2.f);
  float se[NST], sum = 0;
  for (int r = 0; r < NST; ++r) { se[r] = __expf(e[r] * 2.f - m); sum += se[r]; }
  float pn[NST], tot = 0;
  for (int r = 0; r < NST; ++r) { pn[r] = pi_ev[(long)bt * NST + r] * se[r] / sum; tot += pn[r]; }
  float inv = 1.f / fmaxf(tot, 1e-8f);
  float v0 = -1e30f, v1 = -1e30f;
  for (int r = 0; r < NST; ++r) {
    pn[r] *= inv;
    float xx = pn[r];
    if (xx > v0) { v1 = v0; v0 = xx; } else if (xx > v1) v1 = xx;
  }
  float q[NST], qs = 0;
  for (int r = 0; r < NST; ++r) { q[r] = (pn[r] >= v1) ? pn[r] : 0.f; qs += q[r]; }
  float qi = 1.f / fmaxf(qs, 1e-8f);
  for (int r = 0; r < NST; ++r) pi[(long)bt * NST + r] = q[r] * qi;
  g3[bt] = pi[(long)bt * NST + 3];
  g4[bt] = pi[(long)bt * NST + 4];
}
// causal local-window (WIN=4) attention: per-token softmax over offsets
__global__ void k_router_attn(const float* q, const float* kk, const float* v, float* out) {
  __shared__ float red[256];
  __shared__ float a[4];
  int bt = blockIdx.x, t = bt & 255, tid = threadIdx.x;
  float part[4] = {0, 0, 0, 0};
  for (int d = tid; d < DD; d += 256) {
    float qv = q[(long)bt * DD + d];
    for (int o = 0; o < 4; ++o)
      if (t >= o) part[o] += qv * kk[(long)(bt - o) * DD + d];
  }
  float s[4];
  for (int o = 0; o < 4; ++o) s[o] = blockReduce256(part[o], red);
  if (tid == 0) {
    const float scale = 0.03608439182435161f; // 1/sqrt(768)
    float sc[4], m = -1e30f;
    for (int o = 0; o < 4; ++o) { sc[o] = (t >= o) ? s[o] * scale : -1e9f; m = fmaxf(m, sc[o]); }
    float sum = 0;
    for (int o = 0; o < 4; ++o) { sc[o] = __expf(sc[o] - m); sum += sc[o]; }
    for (int o = 0; o < 4; ++o) a[o] = sc[o] / sum;
  }
  __syncthreads();
  for (int d = tid; d < DD; d += 256) {
    float acc = 0;
    for (int o = 0; o < 4; ++o)
      if (t >= o) acc += a[o] * v[(long)(bt - o) * DD + d];
    out[(long)bt * DD + d] = acc;
  }
}
__global__ void k_scale_g(float* x, const float* g) {
  int bt = blockIdx.x;
  float gv = g[bt];
  for (int d = threadIdx.x; d < DD; d += 256) x[(long)bt * DD + d] *= gv;
}
__global__ void k_memctx(const float* scores, const float* mem, float* mem_ctx) {
  __shared__ float a[NSL];
  int bt = blockIdx.x, b = bt >> 8, tid = threadIdx.x;
  if (tid == 0) {
    float m = -1e30f;
    for (int s = 0; s < NSL; ++s) m = fmaxf(m, scores[(long)bt * NSL + s]);
    float sum = 0, e[NSL];
    for (int s = 0; s < NSL; ++s) { e[s] = __expf(scores[(long)bt * NSL + s] - m); sum += e[s]; }
    for (int s = 0; s < NSL; ++s) a[s] = e[s] / sum;
  }
  __syncthreads();
  for (int d = tid; d < DD; d += 256) {
    float acc = 0;
    for (int s = 0; s < NSL; ++s) acc += a[s] * mem[((long)b * NSL + s) * DD + d];
    mem_ctx[(long)bt * DD + d] = acc;
  }
}
__global__ void k_msgs_combine(float* msgs, const float* mem_ctx, const float* g3,
                               const float* pher, float late) {
  int bt = blockIdx.x;
  float c = (0.1f + late * 0.05f * sigmoidf_(pher[bt])) * g3[bt];
  for (int d = threadIdx.x; d < DD; d += 256)
    msgs[(long)bt * DD + d] += c * mem_ctx[(long)bt * DD + d];
}
// precision-weighted write: lm = softplus_raw*g4; mu_new = (lam*mu_w + lm*msgs)/(lam+lm); lam += lm
__global__ void k_write(const float* lam_raw, const float* g4, float* lam,
                        const float* mu_write, const float* msgs, float* mu_new) {
  int bt = blockIdx.x;
  float gv = g4[bt];
  for (int d = threadIdx.x; d < DD; d += 256) {
    long i = (long)bt * DD + d;
    float lm = lam_raw[i] * gv;
    float lo = lam[i];
    mu_new[i] = (lo * mu_write[i] + lm * msgs[i]) / (lo + lm);
    lam[i] = lo + lm;
  }
}
__global__ void k_mu_final(const float* mu_new2, const float* stage_out2, float* mu) {
  int i = blockIdx.x * blockDim.x + threadIdx.x;
  if (i < BTOK * DD) mu[i] = mu_new2[i] + 0.1f * stage_out2[i];
}
__global__ void k_addr_norm(float* addrs, const float* g4) {
  int bt = blockIdx.x * blockDim.x + threadIdx.x;
  if (bt >= BTOK) return;
  float m = -1e30f;
  for (int s = 0; s < NSL; ++s) m = fmaxf(m, addrs[(long)bt * NSL + s]);
  float e[NSL], sum = 0;
  for (int s = 0; s < NSL; ++s) { e[s] = __expf(addrs[(long)bt * NSL + s] - m); sum += e[s]; }
  float gv = g4[bt] / sum;
  for (int s = 0; s < NSL; ++s) addrs[(long)bt * NSL + s] = e[s] * gv;
}
// mem[b,s,:] = 0.9*mem + (1/T) * sum_t addr[b,t,s]*mu[b,t,:]
__global__ void k_mem_update(const float* addrs, const float* mu, float* mem) {
  __shared__ float ap[256];
  int b = blockIdx.x >> 3, s = blockIdx.x & 7, tid = threadIdx.x;
  ap[tid] = addrs[((long)b * 256 + tid) * NSL + s];
  __syncthreads();
  for (int d = tid; d < DD; d += 256) {
    float acc = 0;
    for (int t = 0; t < 256; ++t) acc += ap[t] * mu[((long)b * 256 + t) * DD + d];
    long mi = ((long)b * NSL + s) * DD + d;
    mem[mi] = 0.9f * mem[mi] + acc * (1.f / 256.f);
  }
}
__global__ void k_pher(const float* mu, float* mu_prev, float* pher,
                       const float* g4, float late) {
  __shared__ float red[256];
  int bt = blockIdx.x, tid = threadIdx.x;
  float s = 0;
  for (int d = tid; d < DD; d += 256) {
    float t = mu[(long)bt * DD + d] - mu_prev[(long)bt * DD + d];
    s += t * t;
  }
  float msq = blockReduce256(s, red) * (1.f / DD);
  if (tid == 0) pher[bt] = 0.9f * pher[bt] + late * g4[bt] * tanhf(sqrtf(msq));
  for (int d = tid; d < DD; d += 256)
    mu_prev[(long)bt * DD + d] = mu[(long)bt * DD + d];
}

// ---------------------------------------------------------------------------
struct Pcn { const float *lw, *lb, *gw, *gb, *pe, *m1, *mb1, *m2, *mb2; };
static Pcn getPcn(void* const* din, int base) {
  Pcn p;
  p.lw  = (const float*)din[base + 0];
  p.lb  = (const float*)din[base + 1];
  p.gw  = (const float*)din[base + 2];
  p.gb  = (const float*)din[base + 3];
  p.pe  = (const float*)din[base + 4];
  p.m1  = (const float*)din[base + 5];
  p.mb1 = (const float*)din[base + 6];
  p.m2  = (const float*)din[base + 7];
  p.mb2 = (const float*)din[base + 8];
  return p;
}
static inline void launch_gemm(hipStream_t st, const float* A, const float* B,
                               const float* bias, float* C, int M, int N, int K,
                               int lda, int ldb, int ldc, int Z,
                               long sA, long sB, long sC, long sBias,
                               int transB, int op, float alpha) {
  dim3 grid((N + 63) / 64, (M + 63) / 64, Z), blk(128);
  wmma_gemm<<<grid, blk, 0, st>>>(A, B, bias, C, M, N, K, lda, ldb, ldc,
                                  sA, sB, sC, sBias, transB, op, alpha);
}
static void pcn_apply(hipStream_t st, const float* xin, const Pcn& P,
                      const float* g12, int p, float* gpre, float* out) {
  launch_gemm(st, xin, P.gw, P.gb, gpre, BTOK, DD, DD, DD, DD, DD, 1, 0, 0, 0, 0,
              0, OP_NONE, 1.f);
  k_pcn_finish<<<BTOK, 256, 0, st>>>(xin, gpre, g12, p, P.lw, P.lb, out);
}

extern "C" void kernel_launch(void* const* d_in, const int* in_sizes, int n_in,
                              void* d_out, int out_size, void* d_ws, size_t ws_size,
                              hipStream_t stream) {
  (void)in_sizes; (void)n_in; (void)out_size; (void)ws_size;
  const int*   xids      = (const int*)d_in[0];
  const float* emb       = (const float*)d_in[1];
  const float* pos       = (const float*)d_in[2];
  const float* mu_w      = (const float*)d_in[3];
  const float* mu_b      = (const float*)d_in[4];
  const float* lam_w     = (const float*)d_in[5];
  const float* lam_b     = (const float*)d_in[6];
  const float* tr_w      = (const float*)d_in[7];
  const float* tr_b      = (const float*)d_in[8];
  const float* sb_w1     = (const float*)d_in[9];
  const float* sb_b1     = (const float*)d_in[10];
  const float* sb_w2     = (const float*)d_in[11];
  const float* sb_b2     = (const float*)d_in[12];
  const float* sb_we     = (const float*)d_in[13];
  const float* sb_be     = (const float*)d_in[14];
  const float* wq        = (const float*)d_in[15];
  const float* wk        = (const float*)d_in[16];
  const float* wv        = (const float*)d_in[17];
  const float* wo        = (const float*)d_in[18];
  const float* wl        = (const float*)d_in[19];
  const float* bl        = (const float*)d_in[20];
  const float* slot_keys = (const float*)d_in[21];
  const float* ln_w      = (const float*)d_in[22];
  const float* ln_b      = (const float*)d_in[23];
  Pcn banks[6] = { getPcn(d_in, 24), getPcn(d_in, 33), getPcn(d_in, 42),
                   getPcn(d_in, 51), getPcn(d_in, 60), getPcn(d_in, 69) };

  // workspace bump allocator
  char* base = (char*)d_ws;
  size_t off = 0;
  auto alloc = [&](size_t nf) -> float* {
    float* p = (float*)(base + off);
    off += (nf * sizeof(float) + 255) & ~(size_t)255;
    return p;
  };
  const size_t AD = (size_t)BTOK * DD;
  float* h        = alloc(AD);
  float* mu       = alloc(AD);
  float* lam      = alloc(AD);
  float* mu_prev  = alloc(AD);
  float* pi       = alloc((size_t)BTOK * NST);
  float* pher     = alloc(BTOK);
  float* mem      = alloc((size_t)2 * NSL * DD);
  float* gpre     = alloc(AD);
  float* mu_bank  = alloc(AD);
  float* stage_o  = alloc(AD);
  float* stage_o2 = alloc(AD);
  float* mu_route = alloc(AD);
  float* mu_write = alloc(AD);
  float* msgs0    = alloc(AD);
  float* msgs     = alloc(AD);
  float* trout    = alloc((size_t)BTOK * 36);
  float* pi_ev    = alloc((size_t)BTOK * NST);
  float* h1       = alloc((size_t)NST * BTOK * FFD);
  float* outs     = alloc((size_t)NST * BTOK * DD);
  float* evid     = alloc((size_t)BTOK * NST);
  float* qbuf     = alloc(AD);
  float* kbuf     = alloc(AD);
  float* vbuf     = alloc(AD);
  float* rtmp     = alloc(AD);
  float* mem_ctx  = alloc(AD);
  float* scores   = alloc((size_t)BTOK * NSL);
  float* addrs    = alloc((size_t)BTOK * NSL);
  float* lam_raw  = alloc(AD);
  float* mu_new   = alloc(AD);
  float* mu_new2  = alloc(AD);
  float* fin      = alloc(AD);
  float* g3       = alloc(BTOK);
  float* g4       = alloc(BTOK);
  float* g12[6];
  for (int i = 0; i < 6; ++i) g12[i] = alloc((size_t)12 * 2 * DD);

  const float invsq = 0.03608439182435161f; // 1/sqrt(768)

  // precompute per-pass gate/bias vectors for all 6 PCN banks (tiny MLPs)
  for (int i = 0; i < 6; ++i)
    k_pe_mlp<<<12, 128, 0, stream>>>(banks[i].pe, banks[i].m1, banks[i].mb1,
                                     banks[i].m2, banks[i].mb2, g12[i]);

  // init
  k_embed<<<BTOK, 256, 0, stream>>>(xids, emb, pos, h);
  launch_gemm(stream, h, mu_w, mu_b, mu, BTOK, DD, DD, DD, DD, DD, 1, 0, 0, 0, 0, 0, OP_NONE, 1.f);
  launch_gemm(stream, h, lam_w, lam_b, lam, BTOK, DD, DD, DD, DD, DD, 1, 0, 0, 0, 0, 0, OP_SOFTPLUS01, 1.f);
  k_copy<<<(int)((AD + 255) / 256), 256, 0, stream>>>(mu_prev, mu, (int)AD);
  k_fill<<<(BTOK * NST + 255) / 256, 256, 0, stream>>>(pi, BTOK * NST, 0.f);
  k_pi_init<<<(BTOK + 255) / 256, 256, 0, stream>>>(pi);
  k_fill<<<(BTOK + 255) / 256, 256, 0, stream>>>(pher, BTOK, 0.f);
  k_fill<<<(2 * NSL * DD + 255) / 256, 256, 0, stream>>>(mem, 2 * NSL * DD, 0.f);

  for (int p = 0; p < 12; ++p) {
    float late = (p >= 3) ? 1.f : 0.f;
    // stage transition + pi evolution
    launch_gemm(stream, mu, tr_w, tr_b, trout, BTOK, 36, DD, DD, 36, 36, 1, 0, 0, 0, 0, 0, OP_NONE, 1.f);
    k_trans_pi<<<(BTOK + 255) / 256, 256, 0, stream>>>(trout, pi, pi_ev);
    // stage bank (6 experts, batched WMMA GEMMs)
    pcn_apply(stream, mu, banks[0], g12[0], p, gpre, mu_bank);
    launch_gemm(stream, mu_bank, sb_w1, sb_b1, h1, BTOK, FFD, DD, DD, FFD, FFD,
                NST, 0, (long)DD * FFD, (long)BTOK * FFD, FFD, 0, OP_GELU, 1.f);
    launch_gemm(stream, h1, sb_w2, sb_b2, outs, BTOK, DD, FFD, FFD, DD, DD,
                NST, (long)BTOK * FFD, (long)FFD * DD, (long)BTOK * DD, DD, 0, OP_NONE, 1.f);
    k_stage_mix<<<BTOK, 256, 0, stream>>>(outs, pi, stage_o);
    launch_gemm(stream, mu_bank, sb_we, sb_be, evid, BTOK, NST, DD, DD, NST, NST, 1, 0, 0, 0, 0, 0, OP_NONE, 1.f);
    pcn_apply(stream, stage_o, banks[1], g12[1], p, gpre, stage_o2);
    k_pi_update<<<(BTOK + 255) / 256, 256, 0, stream>>>(pi_ev, evid, pi, g3, g4);
    // router
    pcn_apply(stream, mu, banks[2], g12[2], p, gpre, mu_route);
    launch_gemm(stream, mu_route, wq, nullptr, qbuf, BTOK, DD, DD, DD, DD, DD, 1, 0, 0, 0, 0, 0, OP_NONE, 1.f);
    launch_gemm(stream, mu_route, wk, nullptr, kbuf, BTOK, DD, DD, DD, DD, DD, 1, 0, 0, 0, 0, 0, OP_NONE, 1.f);
    launch_gemm(stream, mu_route, wv, nullptr, vbuf, BTOK, DD, DD, DD, DD, DD, 1, 0, 0, 0, 0, 0, OP_NONE, 1.f);
    k_router_attn<<<BTOK, 256, 0, stream>>>(qbuf, kbuf, vbuf, rtmp);
    launch_gemm(stream, rtmp, wo, nullptr, msgs0, BTOK, DD, DD, DD, DD, DD, 1, 0, 0, 0, 0, 0, OP_NONE, 1.f);
    k_scale_g<<<BTOK, 256, 0, stream>>>(msgs0, g3);
    pcn_apply(stream, msgs0, banks[3], g12[3], p, gpre, msgs);
    // slot-memory attention (batched over B, B-matrix = mem^T)
    launch_gemm(stream, mu, mem, nullptr, scores, 256, NSL, DD, DD, DD, NSL,
                2, (long)256 * DD, (long)NSL * DD, (long)256 * NSL, 0, 1, OP_NONE, invsq);
    k_memctx<<<BTOK, 256, 0, stream>>>(scores, mem, mem_ctx);
    k_msgs_combine<<<BTOK, 256, 0, stream>>>(msgs, mem_ctx, g3, pher, late);
    // precision-weighted write
    pcn_apply(stream, mu, banks[4], g12[4], p, gpre, mu_write);
    launch_gemm(stream, msgs, wl, bl, lam_raw, BTOK, DD, DD, DD, DD, DD, 1, 0, 0, 0, 0, 0, OP_SOFTPLUS, 1.f);
    k_write<<<BTOK, 256, 0, stream>>>(lam_raw, g4, lam, mu_write, msgs, mu_new);
    pcn_apply(stream, mu_new, banks[5], g12[5], p, gpre, mu_new2);
    k_mu_final<<<(int)((AD + 255) / 256), 256, 0, stream>>>(mu_new2, stage_o2, mu);
    // memory write + pheromone
    launch_gemm(stream, mu, slot_keys, nullptr, addrs, BTOK, NSL, DD, DD, DD, NSL, 1, 0, 0, 0, 0, 1, OP_NONE, 1.f);
    k_addr_norm<<<(BTOK + 255) / 256, 256, 0, stream>>>(addrs, g4);
    k_mem_update<<<16, 256, 0, stream>>>(addrs, mu, mem);
    k_pher<<<BTOK, 256, 0, stream>>>(mu, mu_prev, pher, g4, late);
  }

  // final LN + tied-embedding logits (B = emb^T via transB)
  k_layernorm<<<BTOK, 256, 0, stream>>>(mu, ln_w, ln_b, fin);
  launch_gemm(stream, fin, emb, nullptr, (float*)d_out, BTOK, VV, DD, DD, DD, VV,
              1, 0, 0, 0, 0, 1, OP_NONE, invsq);
}
